// NCPCoupling_29781303230573
// MI455X (gfx1250) — matrix operand order repaired and checked by reading, not compiled, not run on hardware
//
#include <hip/hip_runtime.h>
#include <hip/hip_bf16.h>
#include <math.h>

typedef __attribute__((ext_vector_type(16))) _Float16 v16h;
typedef __attribute__((ext_vector_type(8)))  float    v8f;
typedef int v4i __attribute__((vector_size(16)));
typedef __attribute__((address_space(1))) v4i* gv4i_p;   // global AS pointer to int4
typedef __attribute__((address_space(3))) v4i* lv4i_p;   // LDS    AS pointer to int4

#define PI_F 3.14159265358979323846f

constexpr int Bn   = 8192;      // batch
constexpr int Dn   = 1024;      // feature dim
constexpr int Tn   = 512;       // D/2
constexpr int Hn   = 1024;      // hidden (= K for both GEMMs)
constexpr int N2n  = 15360;     // T*EXPR*3
constexpr int EXPn = 10;

#define BK 32
#define LSTR 40                 // padded LDS row stride (halves): bank-spread for b128 reads
#define BN2 240

// ---------------------------------------------------------------------------
// async global->LDS copy (CDNA5 GLOBAL_LOAD_ASYNC_TO_LDS_B128), with fallback
// ---------------------------------------------------------------------------
#define HAS_ASYNC_LDS __has_builtin(__builtin_amdgcn_global_load_async_to_lds_b128)

static __device__ __forceinline__ void copy16_g2l(void* lp, const void* gp) {
#if HAS_ASYNC_LDS
    __builtin_amdgcn_global_load_async_to_lds_b128((gv4i_p)gp, (lv4i_p)lp, 0, 0);
#else
    *(uint4*)lp = *(const uint4*)gp;
#endif
}

static __device__ __forceinline__ void async_wait_all() {
#if HAS_ASYNC_LDS
 #if __has_builtin(__builtin_amdgcn_s_wait_asynccnt)
    __builtin_amdgcn_s_wait_asynccnt(0);
 #else
    asm volatile("s_wait_asynccnt 0" ::: "memory");
 #endif
#endif
}

// ---------------------------------------------------------------------------
// prep: feat = [cos(cond), sin(cond)] (f16), x = tan(0.5*(phi-pi)),
//       copy cond into even columns of z_new
// ---------------------------------------------------------------------------
__global__ __launch_bounds__(256) void prep_kernel(const float* __restrict__ z,
                                                   _Float16* __restrict__ feat,
                                                   float* __restrict__ xbuf,
                                                   float* __restrict__ zout) {
    int idx = blockIdx.x * blockDim.x + threadIdx.x;   // B*T threads
    if (idx >= Bn * Tn) return;
    int b = idx >> 9;
    int t = idx & 511;
    float cond = z[(size_t)b * Dn + 2 * t];
    float phi  = z[(size_t)b * Dn + 2 * t + 1];
    feat[(size_t)b * Dn + t]      = (_Float16)cosf(cond);
    feat[(size_t)b * Dn + Tn + t] = (_Float16)sinf(cond);
    xbuf[idx] = tanf(0.5f * (phi - PI_F));
    zout[(size_t)b * Dn + 2 * t] = cond;
}

// ---------------------------------------------------------------------------
// transpose + f32->f16: dst[n][k] = (f16)src[k][n]  (one-time weight prep)
// ---------------------------------------------------------------------------
__global__ __launch_bounds__(256) void tcvt_kernel(const float* __restrict__ src,
                                                   _Float16* __restrict__ dst,
                                                   int K, int N) {
    __shared__ float tile[32][33];
    int n0 = blockIdx.x * 32;
    int k0 = blockIdx.y * 32;
#pragma unroll
    for (int i = threadIdx.y; i < 32; i += 8)
        tile[i][threadIdx.x] = src[(size_t)(k0 + i) * N + n0 + threadIdx.x];
    __syncthreads();
#pragma unroll
    for (int i = threadIdx.y; i < 32; i += 8)
        dst[(size_t)(n0 + i) * K + k0 + threadIdx.x] = (_Float16)tile[threadIdx.x][i];
}

// ---------------------------------------------------------------------------
// GEMM1 (transposed form): h = relu(feat @ W1 + b1).
// D^T tiles: A = W1t rows (K-contiguous), B = feat rows. 128x128 block, 8 waves.
// Double-buffered async LDS staging; epilogue LDS overlays the tile buffers.
// ---------------------------------------------------------------------------
#define TILE1B (256 * LSTR * 2)          // one (sW 128 + sH 128) buffer, bytes
__global__ __launch_bounds__(256, 1) void gemm1_kernel(const _Float16* __restrict__ feat,
                                                       const _Float16* __restrict__ W1t,
                                                       const float* __restrict__ b1,
                                                       _Float16* __restrict__ hOut) {
    extern __shared__ __align__(16) char smem[];

    const int tid  = threadIdx.x;
    const int wave = tid >> 5;
    const int lane = tid & 31;
    const int m0 = blockIdx.x * 128;     // batch
    const int n0 = blockIdx.y * 128;     // hidden col
    const int K = Hn;

    v8f zero = {};
    v8f acc[8];
#pragma unroll
    for (int i = 0; i < 8; ++i) acc[i] = zero;

    // stage chunk (k0) into buffer buf: sW = W1t rows, sH = feat rows (pure copies)
    auto issue = [&](int k0, int buf) {
        _Float16* sW = (_Float16*)(smem + buf * TILE1B);
        _Float16* sH = sW + 128 * LSTR;
#pragma unroll
        for (int s = 0; s < 2; ++s) {
            int seg = tid + s * 256;
            int row = seg >> 2;
            int c8  = (seg & 3) * 8;
            copy16_g2l(sW + row * LSTR + c8, W1t + (size_t)(n0 + row) * K + k0 + c8);
        }
#pragma unroll
        for (int s = 0; s < 2; ++s) {
            int seg = tid + s * 256;
            int row = seg >> 2;
            int c8  = (seg & 3) * 8;
            copy16_g2l(sH + row * LSTR + c8, feat + (size_t)(m0 + row) * K + k0 + c8);
        }
    };

    issue(0, 0);
    async_wait_all();
    __syncthreads();

    int cur = 0;
    for (int k0 = 0; k0 < K; k0 += BK) {
        if (k0 + BK < K) issue(k0 + BK, cur ^ 1);   // overlap copy(k+1) with compute(k)

        _Float16* sW = (_Float16*)(smem + cur * TILE1B);
        _Float16* sH = sW + 128 * LSTR;

        int brow = wave * 16 + (lane & 15);
        int kb2  = (lane >> 4) * 8;
        v16h bfrag;
        ((uint4*)&bfrag)[0] = *(uint4*)(sH + brow * LSTR + kb2);
        ((uint4*)&bfrag)[1] = *(uint4*)(sH + brow * LSTR + kb2 + 16);

        int kb = (lane >> 4) * 8;
#pragma unroll
        for (int nt = 0; nt < 8; ++nt) {
            int arow = nt * 16 + (lane & 15);
            v16h afrag;
            ((uint4*)&afrag)[0] = *(uint4*)(sW + arow * LSTR + kb);
            ((uint4*)&afrag)[1] = *(uint4*)(sW + arow * LSTR + kb + 16);
            acc[nt] = __builtin_amdgcn_wmma_f32_16x16x32_f16(
                false, afrag, false, bfrag, (short)0, acc[nt], false, false);
        }
        async_wait_all();
        __syncthreads();
        cur ^= 1;
    }

    // epilogue: bias + relu -> f16 -> LDS (overlays tile buffers) -> coalesced store
    _Float16* sC  = (_Float16*)smem;           // 8*16*128 f16 = 32KB, tiles are dead now
    _Float16* myC = sC + wave * 16 * 128;
#pragma unroll
    for (int nt = 0; nt < 8; ++nt) {
#pragma unroll
        for (int r = 0; r < 8; ++r) {
            int col = nt * 16 + ((lane >> 4) << 3) + r;
            int row = lane & 15;
            float v = acc[nt][r] + b1[n0 + col];
            myC[row * 128 + col] = (_Float16)(v > 0.0f ? v : 0.0f);
        }
    }
    __syncthreads();
    for (int s = lane; s < 256; s += 32) {
        int row = s >> 4;
        int c8  = (s & 15) * 8;
        *(uint4*)(hOut + (size_t)(m0 + wave * 16 + row) * Hn + n0 + c8) =
            *(uint4*)(myC + row * 128 + c8);
    }
}

// ---------------------------------------------------------------------------
// GEMM2 fused (transposed form): out = h @ W2 + b2 then softplus/softmax/NCP.
// A = W2t rows (240 cols, 15 wmma tiles), B = h rows (16 per wave).
// Double-buffered async LDS staging; f32 C staging overlays the tile buffers.
// ---------------------------------------------------------------------------
#define TILE2B ((BN2 + 128) * LSTR * 2)      // one (sW 240 + sH 128) buffer, bytes
__global__ __launch_bounds__(256, 1) void gemm2_kernel(const _Float16* __restrict__ hIn,
                                                       const _Float16* __restrict__ W2t,
                                                       const float* __restrict__ b2,
                                                       const float* __restrict__ xbuf,
                                                       float* __restrict__ zout,
                                                       float* __restrict__ partial) {
    extern __shared__ __align__(16) char smem[];

    const int tid  = threadIdx.x;
    const int wave = tid >> 5;
    const int lane = tid & 31;
    const int m0 = blockIdx.x * 128;
    const int n0 = blockIdx.y * BN2;
    const int K = Hn;

    v8f zero = {};
    v8f acc[15];
#pragma unroll
    for (int i = 0; i < 15; ++i) acc[i] = zero;

    auto issue = [&](int k0, int buf) {
        _Float16* sW = (_Float16*)(smem + buf * TILE2B);
        _Float16* sH = sW + BN2 * LSTR;
        for (int seg = tid; seg < 960; seg += 256) {       // 240 rows x 4 segs
            int row = seg >> 2;
            int c8  = (seg & 3) * 8;
            copy16_g2l(sW + row * LSTR + c8, W2t + (size_t)(n0 + row) * K + k0 + c8);
        }
#pragma unroll
        for (int s = 0; s < 2; ++s) {                      // 128 rows x 4 segs
            int seg = tid + s * 256;
            int row = seg >> 2;
            int c8  = (seg & 3) * 8;
            copy16_g2l(sH + row * LSTR + c8, hIn + (size_t)(m0 + row) * K + k0 + c8);
        }
    };

    issue(0, 0);
    async_wait_all();
    __syncthreads();

    int cur = 0;
    for (int k0 = 0; k0 < K; k0 += BK) {
        if (k0 + BK < K) issue(k0 + BK, cur ^ 1);   // overlap copy(k+1) with compute(k)

        _Float16* sW = (_Float16*)(smem + cur * TILE2B);
        _Float16* sH = sW + BN2 * LSTR;

        int brow = wave * 16 + (lane & 15);
        int kb2  = (lane >> 4) * 8;
        v16h bfrag;
        ((uint4*)&bfrag)[0] = *(uint4*)(sH + brow * LSTR + kb2);
        ((uint4*)&bfrag)[1] = *(uint4*)(sH + brow * LSTR + kb2 + 16);

        int kb = (lane >> 4) * 8;
#pragma unroll
        for (int nt = 0; nt < 15; ++nt) {
            int arow = nt * 16 + (lane & 15);
            v16h afrag;
            ((uint4*)&afrag)[0] = *(uint4*)(sW + arow * LSTR + kb);
            ((uint4*)&afrag)[1] = *(uint4*)(sW + arow * LSTR + kb + 16);
            acc[nt] = __builtin_amdgcn_wmma_f32_16x16x32_f16(
                false, afrag, false, bfrag, (short)0, acc[nt], false, false);
        }
        async_wait_all();
        __syncthreads();
        cur ^= 1;
    }

    // stage D^T tile in LDS as [batch-local][col] (overlays dead tile buffers)
    float* sC  = (float*)smem;                 // 8*16*240 f32 = 120KB
    float* myC = sC + wave * 16 * BN2;
#pragma unroll
    for (int nt = 0; nt < 15; ++nt) {
#pragma unroll
        for (int r = 0; r < 8; ++r) {
            int col = nt * 16 + ((lane >> 4) << 3) + r;
            int row = lane & 15;
            myC[row * BN2 + col] = acc[nt][r];
        }
    }
    __syncthreads();

    // fused NCP epilogue: lane l -> row = l/2, 4 t-groups (l%2)*4 + 0..3
    int row = lane >> 1;
    int b   = m0 + wave * 16 + row;
    float ldsum = 0.0f;
    for (int s = 0; s < 4; ++s) {
        int tg = (lane & 1) * 4 + s;
        int t  = blockIdx.y * 8 + tg;            // global t in [0,512)
        const float* v = myC + row * BN2 + tg * 30;
        float w[30];
#pragma unroll
        for (int j = 0; j < 30; ++j) w[j] = v[j] + b2[n0 + tg * 30 + j];

        // softmax over rho logits w[3e+2]
        float mx = -1e30f;
#pragma unroll
        for (int e = 0; e < EXPn; ++e) mx = fmaxf(mx, w[3 * e + 2]);
        float re[EXPn];
        float se = 0.0f;
#pragma unroll
        for (int e = 0; e < EXPn; ++e) { re[e] = __expf(w[3 * e + 2] - mx); se += re[e]; }
        float inv = 1.0f / se;

        float x  = xbuf[(size_t)b * Tn + t];
        float xx = 1.0f + x * x;
        float osum = 0.0f, dsum = 0.0f;
#pragma unroll
        for (int e = 0; e < EXPn; ++e) {
            float a0 = w[3 * e];
            float alpha = (a0 > 20.0f ? a0 : log1pf(__expf(a0))) + 0.001f;
            float beta  = w[3 * e + 1];
            float u   = alpha * x + beta;
            float rho = re[e] * inv;
            osum += rho * (2.0f * atanf(u) + PI_F);
            dsum += rho * alpha * xx / (1.0f + u * u);
        }
        zout[(size_t)b * Dn + 2 * t + 1] = osum;
        ldsum += __logf(dsum);
    }
    float other = __shfl_xor(ldsum, 1, 32);
    if ((lane & 1) == 0)
        partial[(size_t)b * 64 + blockIdx.y] = ldsum + other;
}

// ---------------------------------------------------------------------------
// reduce partials -> log_det
// ---------------------------------------------------------------------------
__global__ __launch_bounds__(256) void reduce_kernel(const float* __restrict__ partial,
                                                     float* __restrict__ logdet) {
    int b = blockIdx.x * blockDim.x + threadIdx.x;
    if (b >= Bn) return;
    float s = 0.0f;
#pragma unroll 8
    for (int i = 0; i < 64; ++i) s += partial[(size_t)b * 64 + i];
    logdet[b] = s;
}

// ---------------------------------------------------------------------------
extern "C" void kernel_launch(void* const* d_in, const int* in_sizes, int n_in,
                              void* d_out, int out_size, void* d_ws, size_t ws_size,
                              hipStream_t stream) {
    const float* z  = (const float*)d_in[0];
    const float* W1 = (const float*)d_in[1];
    const float* b1 = (const float*)d_in[2];
    const float* W2 = (const float*)d_in[3];
    const float* b2 = (const float*)d_in[4];

    float* zout   = (float*)d_out;               // B*D z_new
    float* logdet = zout + (size_t)Bn * Dn;      // B   log_det

    char* ws = (char*)d_ws;
    _Float16* featH = (_Float16*)ws;  ws += (size_t)Bn * Dn * 2;     // 16MB
    _Float16* hH    = (_Float16*)ws;  ws += (size_t)Bn * Hn * 2;     // 16MB
    _Float16* W1t   = (_Float16*)ws;  ws += (size_t)Hn * Hn * 2;     // 2MB  (N x K)
    _Float16* W2t   = (_Float16*)ws;  ws += (size_t)Hn * N2n * 2;    // 30MB (N x K)
    float*    xbuf  = (float*)ws;     ws += (size_t)Bn * Tn * 4;     // 16MB
    float*    part  = (float*)ws;     ws += (size_t)Bn * 64 * 4;     // 2MB

    // 1) prep
    prep_kernel<<<(Bn * Tn + 255) / 256, 256, 0, stream>>>(z, featH, xbuf, zout);

    // 2) one-time weight transpose + f16 conversion
    {
        dim3 blk(32, 8);
        tcvt_kernel<<<dim3(Hn / 32, Hn / 32), blk, 0, stream>>>(W1, W1t, Hn, Hn);
        tcvt_kernel<<<dim3(N2n / 32, Hn / 32), blk, 0, stream>>>(W2, W2t, Hn, N2n);
    }

    // 3) GEMM1: h = relu(feat @ W1 + b1)
    {
        dim3 grid(Bn / 128, Hn / 128);
        size_t shmem = 2 * TILE1B;                                   // 40KB (>= 32KB sC)
        if (shmem < (size_t)8 * 16 * 128 * 2) shmem = (size_t)8 * 16 * 128 * 2;
        gemm1_kernel<<<grid, 256, shmem, stream>>>(featH, W1t, b1, hH);
    }

    // 4) GEMM2 fused with NCP epilogue
    {
        dim3 grid(Bn / 128, N2n / BN2);  // 64 x 64
        size_t shmem = (size_t)8 * 16 * BN2 * 4;                     // 120KB (>= 2*TILE2B)
        if (shmem < (size_t)2 * TILE2B) shmem = (size_t)2 * TILE2B;
        gemm2_kernel<<<grid, 256, shmem, stream>>>(hH, W2t, b2, xbuf, zout, part);
    }

    // 5) log_det reduction
    reduce_kernel<<<(Bn + 255) / 256, 256, 0, stream>>>(part, logdet);
}
